// GNNEmbVariableEncoder_88502096101407
// MI455X (gfx1250) — matrix operand order repaired and checked by reading, not compiled, not run on hardware
//
#include <hip/hip_runtime.h>
#include <math.h>

typedef __attribute__((ext_vector_type(16))) _Float16 v16h;
typedef __attribute__((ext_vector_type(8)))  _Float16 v8h;
typedef __attribute__((ext_vector_type(8)))  float    v8f;

#define Bn 1024   // batch
#define Ln 256    // max group length
#define Tn 32     // temp dim
#define Hn 1024   // hidden
#define Kn 1056   // T*(T+1) = concat embedding width

// ---------------------------------------------------------------------------
// Kernel 1: per-row masked sums.  agg[b,:] = s[b]*W + len[b]*bias, so all we
// need per row is s[b] = sum_{t<len} data[b,t] and len[b].  One wave32 per
// row, 8 rows per 256-thread block.  Reads weight+bias once (2 MB total).
// ---------------------------------------------------------------------------
__global__ void row_stats_kernel(const float* __restrict__ weight,
                                 const float* __restrict__ bias,
                                 const int*   __restrict__ wlen,
                                 const int*   __restrict__ blen,
                                 float4*      __restrict__ stats) {
    const int wave = threadIdx.x >> 5;
    const int lane = threadIdx.x & 31;
    const int b    = blockIdx.x * 8 + wave;

    const int lw = wlen[b];
    const int lb = blen[b];

    float sw = 0.0f, sb = 0.0f;
#pragma unroll
    for (int i = 0; i < Ln / 32; ++i) {
        const int t = lane + 32 * i;
        const float w = weight[b * Ln + t];
        const float x = bias[b * Ln + t];
        sw += (t < lw) ? w : 0.0f;
        sb += (t < lb) ? x : 0.0f;
    }
#pragma unroll
    for (int off = 16; off > 0; off >>= 1) {
        sw += __shfl_xor(sw, off, 32);
        sb += __shfl_xor(sb, off, 32);
    }
    if (lane == 0) stats[b] = make_float4(sw, (float)lw, sb, (float)lb);
}

// ---------------------------------------------------------------------------
// Kernel 2: emb[b,k] = sigmoid(s*W[k] + len*bias[k]), stored as f16 for WMMA.
// k < 1024 -> weight transformer, k >= 1024 -> bias transformer.
// ---------------------------------------------------------------------------
__global__ void build_emb_kernel(const float4* __restrict__ stats,
                                 const float* __restrict__ Ww, const float* __restrict__ bw,
                                 const float* __restrict__ Wb, const float* __restrict__ bb,
                                 _Float16* __restrict__ emb) {
    const int idx = blockIdx.x * blockDim.x + threadIdx.x;
    const int b = idx / Kn;
    const int k = idx - b * Kn;
    const float4 s = stats[b];
    float v;
    if (k < Tn * Tn) v = s.x * Ww[k] + s.y * bw[k];
    else             v = s.z * Wb[k - Tn * Tn] + s.w * bb[k - Tn * Tn];
    v = 1.0f / (1.0f + __expf(-v));
    emb[idx] = (_Float16)v;
}

// ---------------------------------------------------------------------------
// Kernel 3: We f32 -> f16 (one-time convert; 2.1 MB result stays in L2).
// ---------------------------------------------------------------------------
__global__ void cvt_we_kernel(const float* __restrict__ We,
                              _Float16* __restrict__ We16) {
    const int idx = blockIdx.x * blockDim.x + threadIdx.x;
    We16[idx] = (_Float16)We[idx];
}

// ---------------------------------------------------------------------------
// Kernel 4: out = relu(emb @ We.T + be) via v_wmma_f32_16x16x32_f16.
// One wave per 16x16 output tile; 8 waves (8 n-tiles) per block.
// A fragment (16x32 f16, row-major emb):
//   lane l: m = l&15; halves 0..7  = emb[m, kb*32 + (l>>4)*8 .. +7]
//                     halves 8..15 = emb[m, kb*32 + 16 + (l>>4)*8 .. +7]
// B fragment (32x16 f16, B[k,n] = We[n,k], We row-major => contiguous in k):
//   lane l: n = l&15; halves 0..7 = We[h0+n, kb*32 + (l>>4)*8 .. +7], halves 8..15 = We[h0+n, kb*32 + 16 + (l>>4)*8 .. +7]
// Both are plain contiguous 128-bit global loads (global_load_b128).
// ---------------------------------------------------------------------------
__global__ void wmma_gemm_kernel(const _Float16* __restrict__ emb,
                                 const _Float16* __restrict__ We16,
                                 const float*    __restrict__ be,
                                 float*          __restrict__ out) {
    const int lane = threadIdx.x & 31;
    const int wave = threadIdx.x >> 5;
    const int nt   = blockIdx.x * 8 + wave;   // n-tile 0..63
    const int mt   = blockIdx.y;              // m-tile 0..63
    const int b0   = mt * 16;
    const int h0   = nt * 16;
    const int mn   = lane & 15;               // row for A, col for B
    const int hi   = lane >> 4;               // half-wave selector

    const _Float16* arow = emb  + (size_t)(b0 + mn) * Kn + hi * 8;
    const _Float16* brow = We16 + (size_t)(h0 + mn) * Kn + hi * 8;

    v8f c = {};
#pragma unroll 2
    for (int kb = 0; kb < Kn / 32; ++kb) {    // 33 K-steps of 32
        const v8h a0 = *(const v8h*)(arow + kb * 32);
        const v8h a1 = *(const v8h*)(arow + kb * 32 + 16);
        const v8h bl = *(const v8h*)(brow + kb * 32);
        const v8h bh = *(const v8h*)(brow + kb * 32 + 16);
        const v16h a = __builtin_shufflevector(a0, a1,
            0,1,2,3,4,5,6,7,8,9,10,11,12,13,14,15);
        const v16h b = __builtin_shufflevector(bl, bh,
            0,1,2,3,4,5,6,7,8,9,10,11,12,13,14,15);
        // (neg_a, A, neg_b, B, c_mod, C, reuse_a, reuse_b)
        c = __builtin_amdgcn_wmma_f32_16x16x32_f16(
                false, a, false, b, (short)0, c, false, false);
    }

    // C/D layout: VGPR r -> M = r (lanes 0-15) or 8+r (lanes 16-31), N = lane&15
    const int col = h0 + mn;
    const float bias_h = be[col];
#pragma unroll
    for (int r = 0; r < 8; ++r) {
        const int row = b0 + r + hi * 8;
        float v = c[r] + bias_h;
        out[(size_t)row * Hn + col] = v > 0.0f ? v : 0.0f;
    }
}

// ---------------------------------------------------------------------------
// Launch.  Workspace layout:
//   [0, 16 KB)                 float4 stats[1024]
//   [16 KB, +2,162,688)        _Float16 emb[1024*1056]
//   [.., +2,162,688)           _Float16 We16[1024*1056]
// Total ~4.3 MB.
// ---------------------------------------------------------------------------
extern "C" void kernel_launch(void* const* d_in, const int* in_sizes, int n_in,
                              void* d_out, int out_size, void* d_ws, size_t ws_size,
                              hipStream_t stream) {
    const float* weight = (const float*)d_in[0];
    const float* bias   = (const float*)d_in[1];
    const int*   wlen   = (const int*)d_in[2];
    const int*   blen   = (const int*)d_in[3];
    const float* Ww     = (const float*)d_in[4];
    const float* bw     = (const float*)d_in[5];
    const float* Wb     = (const float*)d_in[6];
    const float* bb     = (const float*)d_in[7];
    const float* We     = (const float*)d_in[8];
    const float* be     = (const float*)d_in[9];
    float* out = (float*)d_out;

    char* ws = (char*)d_ws;
    float4*   stats = (float4*)ws;
    _Float16* emb   = (_Float16*)(ws + 16384);
    _Float16* We16  = (_Float16*)(ws + 16384 + (size_t)Bn * Kn * sizeof(_Float16));

    row_stats_kernel<<<Bn / 8, 256, 0, stream>>>(weight, bias, wlen, blen, stats);
    build_emb_kernel<<<(Bn * Kn) / 256, 256, 0, stream>>>(stats, Ww, bw, Wb, bb, emb);
    cvt_we_kernel<<<(Hn * Kn) / 256, 256, 0, stream>>>(We, We16);
    wmma_gemm_kernel<<<dim3(8, 64), 256, 0, stream>>>(emb, We16, be, out);
}
